// KQAlignedAttention_22874995819080
// MI455X (gfx1250) — compile-verified, matrix operand order checked by reading, not run on hardware
//
#include <hip/hip_runtime.h>

// ---------------------------------------------------------------------------
// KQ-aligned attention for MI455X (gfx1250, wave32, WMMA bf16 16x16x32).
//
// T1+T2 = (M1_flat @ Wq) + (M2_flat @ Wk)
//   M1 = softmax(beta QK^T) @ K   (per b,h)
//   M2 = softmax(beta QK^T)^T @ Q (per b,h)
// Transposed copies (Q^T, K^T, W^T) are written at production time so every
// WMMA B-fragment is two contiguous 16B loads (no LDS transposes).
// Tile staging uses GLOBAL_LOAD_ASYNC_TO_LDS_B128 (ASYNCcnt) when available.
// ---------------------------------------------------------------------------

typedef __attribute__((ext_vector_type(16))) __bf16 v16bf;
typedef __attribute__((ext_vector_type(8)))  __bf16 v8bf;
typedef __attribute__((ext_vector_type(8)))  float  v8f;
typedef __attribute__((ext_vector_type(4)))  int    v4i;

#define BB    8
#define NN    1024
#define DIMM  768
#define HH    12
#define HDD   64
#define MROWS (BB * NN)   // 8192

#if defined(__HIP_DEVICE_COMPILE__) && \
    __has_builtin(__builtin_amdgcn_global_load_async_to_lds_b128) && \
    __has_builtin(__builtin_amdgcn_s_wait_asynccnt)
#define USE_ASYNC 1
#else
#define USE_ASYNC 0
#endif

// 16B global -> LDS copy: async (ASYNCcnt) when available, else reg round-trip.
// Builtin prototype (from compiler diagnostic): (v4i AS1*, v4i AS3*, Ii, Ii).
__device__ __forceinline__ void g2l_16B(__bf16* l, const __bf16* g) {
#if USE_ASYNC
  __builtin_amdgcn_global_load_async_to_lds_b128(
      (__attribute__((address_space(1))) v4i*)g,
      (__attribute__((address_space(3))) v4i*)l, 0, 0);
#else
  *(v8bf*)l = *(const v8bf*)g;
#endif
}

__device__ __forceinline__ void wait_g2l() {
#if USE_ASYNC
  __builtin_amdgcn_s_wait_asynccnt(0);
#endif
}

// 16x32 bf16 WMMA fragment (A-layout, or B-layout from an N-major source).
// Per CDNA5 ISA 7.12.2: lane L -> row = L%16, half = L/16;
// elems 0..7 = src[row][col + hf*8 ..], elems 8..15 = src[row][col+16+hf*8 ..]
__device__ __forceinline__ v16bf load_frag(const __bf16* base, int stride,
                                           int row, int col, int hf) {
  const __bf16* p = base + (size_t)row * stride + col + hf * 8;
  v8bf lo = *(const v8bf*)p;
  v8bf hi = *(const v8bf*)(p + 16);
  v16bf r;
#pragma unroll
  for (int i = 0; i < 8; ++i) { r[i] = lo[i]; r[i + 8] = hi[i]; }
  return r;
}

__device__ __forceinline__ v8f wmma_bf16(v16bf a, v16bf b, v8f c) {
  return __builtin_amdgcn_wmma_f32_16x16x32_bf16(false, a, false, b,
                                                 (short)0, c, false, false);
}

// ---------------------------------------------------------------------------
__global__ __launch_bounds__(256) void cvt_bf16(const float* __restrict__ src,
                                                __bf16* __restrict__ dst, int n) {
  int i = blockIdx.x * blockDim.x + threadIdx.x;
  if (i < n) dst[i] = (__bf16)src[i];
}

// Weights: emit both [e][d] (bf16) and transposed [d][e] (bf16).
__global__ __launch_bounds__(256) void cvt_w(const float* __restrict__ src,
                                             __bf16* __restrict__ w,
                                             __bf16* __restrict__ wt) {
  int i = blockIdx.x * blockDim.x + threadIdx.x;
  if (i < DIMM * DIMM) {
    __bf16 b = (__bf16)src[i];
    int e = i / DIMM, d = i - e * DIMM;
    w[i] = b;
    wt[(size_t)d * DIMM + e] = b;
  }
}

// ---------------------------------------------------------------------------
// C[m,e] = sum_d A[m,d] * W[e,d]  (A @ W^T), bf16 out in per-head layout
// Dst [bh][n][z] and transposed DstT [bh][z][n].
__global__ __launch_bounds__(128) void gemm_proj(const __bf16* __restrict__ A,
                                                 const __bf16* __restrict__ W,
                                                 __bf16* __restrict__ Dst,
                                                 __bf16* __restrict__ DstT) {
  __shared__ __align__(16) __bf16 As[64][40];
  __shared__ __align__(16) __bf16 Bs[64][40];
  const int t = threadIdx.x, lane = t & 31, w = t >> 5;
  const int ln = lane & 15, hf = lane >> 4;
  const int mBase = blockIdx.x * 64, eBase = blockIdx.y * 64;
  v8f acc[4] = {};
  for (int k0 = 0; k0 < DIMM; k0 += 32) {
#pragma unroll
    for (int i = 0; i < 2; ++i) {          // 64x32 tile: 256 x 16B vectors
      int v = t * 2 + i;
      int r = v >> 2, s = (v & 3) * 8;
      g2l_16B(&As[r][s], &A[(size_t)(mBase + r) * DIMM + k0 + s]);
      g2l_16B(&Bs[r][s], &W[(size_t)(eBase + r) * DIMM + k0 + s]);
    }
    if (k0 + 32 < DIMM)
      __builtin_prefetch(&A[(size_t)(mBase + (t >> 1)) * DIMM + k0 + 32], 0, 1);
    wait_g2l();
    __syncthreads();
    v16bf a = load_frag(&As[0][0], 40, w * 16 + ln, 0, hf);
#pragma unroll
    for (int j = 0; j < 4; ++j) {
      v16bf b = load_frag(&Bs[0][0], 40, j * 16 + ln, 0, hf);
      acc[j] = wmma_bf16(a, b, acc[j]);
    }
    __syncthreads();
  }
#pragma unroll
  for (int j = 0; j < 4; ++j)
#pragma unroll
    for (int r = 0; r < 8; ++r) {
      int m = mBase + w * 16 + r + 8 * hf;
      int e = eBase + j * 16 + ln;
      int b = m >> 10, n = m & 1023;
      int h = e >> 6, z = e & 63;
      int bh = b * HH + h;
      __bf16 val = (__bf16)acc[j][r];
      Dst[(((size_t)bh << 10) + n) * HDD + z] = val;
      DstT[((size_t)bh * HDD + z) * NN + n] = val;
    }
}

// ---------------------------------------------------------------------------
// Pass 1: per-(b,h,q) softmax row max & sum(exp) over all k. No LDS.
__global__ __launch_bounds__(128) void attn_stats(const __bf16* __restrict__ Qb,
                                                  const __bf16* __restrict__ Kb,
                                                  const float* __restrict__ betas,
                                                  float* __restrict__ rowmax,
                                                  float* __restrict__ rowsum) {
  const int t = threadIdx.x, lane = t & 31, w = t >> 5;
  const int ln = lane & 15, hf = lane >> 4;
  const int bh = blockIdx.y, qBase = blockIdx.x * 64;
  const float beta = betas[bh % HH];
  const __bf16* Q = Qb + (size_t)bh * NN * HDD;
  const __bf16* K = Kb + (size_t)bh * NN * HDD;
  v16bf qa0 = load_frag(Q, HDD, qBase + w * 16 + ln, 0, hf);
  v16bf qa1 = load_frag(Q, HDD, qBase + w * 16 + ln, 32, hf);
  float mrow[8], srow[8];
#pragma unroll
  for (int r = 0; r < 8; ++r) { mrow[r] = -3.0e38f; srow[r] = 0.0f; }
  for (int k0 = 0; k0 < NN; k0 += 16) {
    v8f s = {};
    v16bf b0 = load_frag(K, HDD, k0 + ln, 0, hf);
    v16bf b1 = load_frag(K, HDD, k0 + ln, 32, hf);
    s = wmma_bf16(qa0, b0, s);
    s = wmma_bf16(qa1, b1, s);
#pragma unroll
    for (int r = 0; r < 8; ++r) {
      float v = s[r] * beta;
      float tm = v;
      tm = fmaxf(tm, __shfl_xor(tm, 1, 32));
      tm = fmaxf(tm, __shfl_xor(tm, 2, 32));
      tm = fmaxf(tm, __shfl_xor(tm, 4, 32));
      tm = fmaxf(tm, __shfl_xor(tm, 8, 32));
      float nm = fmaxf(mrow[r], tm);
      float e = __expf(v - nm);
      e += __shfl_xor(e, 1, 32);
      e += __shfl_xor(e, 2, 32);
      e += __shfl_xor(e, 4, 32);
      e += __shfl_xor(e, 8, 32);
      srow[r] = srow[r] * __expf(mrow[r] - nm) + e;
      mrow[r] = nm;
    }
  }
  if (ln == 0) {
#pragma unroll
    for (int r = 0; r < 8; ++r) {
      int q = qBase + w * 16 + r + 8 * hf;
      rowmax[bh * NN + q] = mrow[r];
      rowsum[bh * NN + q] = srow[r];
    }
  }
}

// ---------------------------------------------------------------------------
// M1 = P @ K (parallel over q-blocks). LDS only for the wave-private
// C-layout -> A-layout shuffle of P; no barriers needed.
__global__ __launch_bounds__(128) void attn_m1(const __bf16* __restrict__ Qb,
                                               const __bf16* __restrict__ Kb,
                                               const __bf16* __restrict__ KT,
                                               const float* __restrict__ betas,
                                               const float* __restrict__ rowmax,
                                               const float* __restrict__ rowsum,
                                               __bf16* __restrict__ M1) {
  __shared__ __align__(16) __bf16 Ps[64][40];   // P [q][k] bf16, wave-private rows
  const int t = threadIdx.x, lane = t & 31, w = t >> 5;
  const int ln = lane & 15, hf = lane >> 4;
  const int bh = blockIdx.y, qBase = blockIdx.x * 64;
  const int b = bh / HH, h = bh % HH;
  const float beta = betas[h];
  const __bf16* Q  = Qb + (size_t)bh * NN * HDD;
  const __bf16* K  = Kb + (size_t)bh * NN * HDD;
  const __bf16* Kt = KT + (size_t)bh * HDD * NN;  // [z][n]
  v16bf qa0 = load_frag(Q, HDD, qBase + w * 16 + ln, 0, hf);
  v16bf qa1 = load_frag(Q, HDD, qBase + w * 16 + ln, 32, hf);
  float pm[8], pinv[8];
#pragma unroll
  for (int r = 0; r < 8; ++r) {
    int q = qBase + w * 16 + r + 8 * hf;
    pm[r]   = rowmax[bh * NN + q];
    pinv[r] = 1.0f / rowsum[bh * NN + q];
  }
  v8f accm[4] = {};
  for (int k0 = 0; k0 < NN; k0 += 32) {
#pragma unroll
    for (int kc = 0; kc < 32; kc += 16) {
      v8f sacc = {};
      v16bf b0 = load_frag(K, HDD, k0 + kc + ln, 0, hf);
      v16bf b1 = load_frag(K, HDD, k0 + kc + ln, 32, hf);
      sacc = wmma_bf16(qa0, b0, sacc);
      sacc = wmma_bf16(qa1, b1, sacc);
#pragma unroll
      for (int r = 0; r < 8; ++r) {
        float p = __expf(sacc[r] * beta - pm[r]) * pinv[r];
        Ps[w * 16 + r + 8 * hf][kc + ln] = (__bf16)p;
      }
    }
    v16bf aP = load_frag(&Ps[0][0], 40, w * 16 + ln, 0, hf);
#pragma unroll
    for (int j = 0; j < 4; ++j) {
      v16bf bK = load_frag(Kt, NN, j * 16 + ln, k0, hf);  // B^T[z][k]
      accm[j] = wmma_bf16(aP, bK, accm[j]);
    }
  }
#pragma unroll
  for (int j = 0; j < 4; ++j)
#pragma unroll
    for (int r = 0; r < 8; ++r) {
      int n = qBase + w * 16 + r + 8 * hf;
      M1[((size_t)(b * NN + n)) * DIMM + h * HDD + j * 16 + ln] = (__bf16)accm[j][r];
    }
}

// ---------------------------------------------------------------------------
// M2 = P^T @ Q (parallel over k-blocks, loop over q). Barrier-free.
__global__ __launch_bounds__(128) void attn_m2(const __bf16* __restrict__ Qb,
                                               const __bf16* __restrict__ Kb,
                                               const __bf16* __restrict__ QT,
                                               const float* __restrict__ betas,
                                               const float* __restrict__ rowmax,
                                               const float* __restrict__ rowsum,
                                               __bf16* __restrict__ M2) {
  __shared__ __align__(16) __bf16 Pt[64][40];   // P^T [k][q], wave-private rows
  const int t = threadIdx.x, lane = t & 31, w = t >> 5;
  const int ln = lane & 15, hf = lane >> 4;
  const int bh = blockIdx.y, kBase = blockIdx.x * 64;
  const int b = bh / HH, h = bh % HH;
  const float beta = betas[h];
  const __bf16* Q  = Qb + (size_t)bh * NN * HDD;
  const __bf16* K  = Kb + (size_t)bh * NN * HDD;
  const __bf16* Qt = QT + (size_t)bh * HDD * NN;  // [z][n]
  v16bf kb0 = load_frag(K, HDD, kBase + w * 16 + ln, 0, hf);
  v16bf kb1 = load_frag(K, HDD, kBase + w * 16 + ln, 32, hf);
  v8f accm[4] = {};
  for (int q0 = 0; q0 < NN; q0 += 32) {
#pragma unroll
    for (int qc = 0; qc < 32; qc += 16) {
      v8f sacc = {};
      v16bf a0 = load_frag(Q, HDD, q0 + qc + ln, 0, hf);
      v16bf a1 = load_frag(Q, HDD, q0 + qc + ln, 32, hf);
      sacc = wmma_bf16(a0, kb0, sacc);
      sacc = wmma_bf16(a1, kb1, sacc);
#pragma unroll
      for (int r = 0; r < 8; ++r) {
        int q = q0 + qc + r + 8 * hf;
        float p = __expf(sacc[r] * beta - rowmax[bh * NN + q]) / rowsum[bh * NN + q];
        Pt[w * 16 + ln][qc + r + 8 * hf] = (__bf16)p;
      }
    }
    v16bf aPt = load_frag(&Pt[0][0], 40, w * 16 + ln, 0, hf);
#pragma unroll
    for (int j = 0; j < 4; ++j) {
      v16bf bQ = load_frag(Qt, NN, j * 16 + ln, q0, hf);  // B^T[z][q]
      accm[j] = wmma_bf16(aPt, bQ, accm[j]);
    }
  }
#pragma unroll
  for (int j = 0; j < 4; ++j)
#pragma unroll
    for (int r = 0; r < 8; ++r) {
      int n = kBase + w * 16 + r + 8 * hf;
      M2[((size_t)(b * NN + n)) * DIMM + h * HDD + j * 16 + ln] = (__bf16)accm[j][r];
    }
}

// ---------------------------------------------------------------------------
// Out[m,d] = sum_e M1[m,e]*Wq[e,d] + M2[m,e]*Wk[e,d]  (f32 out).
// B tiles from pre-transposed W^T [d][e]: pure copies -> async staging.
__global__ __launch_bounds__(128) void gemm_out(const __bf16* __restrict__ M1,
                                                const __bf16* __restrict__ M2,
                                                const __bf16* __restrict__ WqT,
                                                const __bf16* __restrict__ WkT,
                                                float* __restrict__ Out) {
  __shared__ __align__(16) __bf16 BqS[64][40];  // [d][e] tile
  __shared__ __align__(16) __bf16 BkS[64][40];
  const int t = threadIdx.x, lane = t & 31, w = t >> 5;
  const int ln = lane & 15, hf = lane >> 4;
  const int mBase = blockIdx.x * 64, dBase = blockIdx.y * 64;
  v8f acc[4] = {};
  for (int e0 = 0; e0 < DIMM; e0 += 32) {
#pragma unroll
    for (int i = 0; i < 2; ++i) {             // 64x32 tiles: 256 x 16B vectors
      int v = t * 2 + i;
      int r = v >> 2, s = (v & 3) * 8;
      g2l_16B(&BqS[r][s], &WqT[(size_t)(dBase + r) * DIMM + e0 + s]);
      g2l_16B(&BkS[r][s], &WkT[(size_t)(dBase + r) * DIMM + e0 + s]);
    }
    if (e0 + 32 < DIMM)
      __builtin_prefetch(&M1[(size_t)(mBase + (t >> 1)) * DIMM + e0 + 32], 0, 1);
    wait_g2l();
    __syncthreads();
    v16bf a1 = load_frag(M1, DIMM, mBase + w * 16 + ln, e0, hf);
    v16bf a2 = load_frag(M2, DIMM, mBase + w * 16 + ln, e0, hf);
#pragma unroll
    for (int j = 0; j < 4; ++j) {
      v16bf bq = load_frag(&BqS[0][0], 40, j * 16 + ln, 0, hf);
      v16bf bk = load_frag(&BkS[0][0], 40, j * 16 + ln, 0, hf);
      acc[j] = wmma_bf16(a1, bq, acc[j]);
      acc[j] = wmma_bf16(a2, bk, acc[j]);
    }
    __syncthreads();
  }
#pragma unroll
  for (int j = 0; j < 4; ++j)
#pragma unroll
    for (int r = 0; r < 8; ++r) {
      int m = mBase + w * 16 + r + 8 * hf;
      Out[(size_t)m * DIMM + dBase + j * 16 + ln] = acc[j][r];
    }
}

// ---------------------------------------------------------------------------
extern "C" void kernel_launch(void* const* d_in, const int* in_sizes, int n_in,
                              void* d_out, int out_size, void* d_ws, size_t ws_size,
                              hipStream_t stream) {
  const float* x     = (const float*)d_in[0];   // [8,1024,768]
  const float* Wq    = (const float*)d_in[1];   // [768,768]
  const float* Wk    = (const float*)d_in[2];   // [768,768]
  const float* betas = (const float*)d_in[3];   // [12]
  float* out = (float*)d_out;                   // [8192,768] f32

  char* p = (char*)d_ws;
  auto carve = [&](size_t bytes) -> char* {
    char* r = p;
    p += (bytes + 255) & ~(size_t)255;
    return r;
  };
  const size_t xN = (size_t)MROWS * DIMM;       // 6.29M elems
  const size_t wN = (size_t)DIMM * DIMM;
  __bf16* xb   = (__bf16*)carve(xN * 2);
  __bf16* Wqb  = (__bf16*)carve(wN * 2);
  __bf16* Wkb  = (__bf16*)carve(wN * 2);
  __bf16* WqT  = (__bf16*)carve(wN * 2);
  __bf16* WkT  = (__bf16*)carve(wN * 2);
  __bf16* Qb   = (__bf16*)carve(xN * 2);        // [bh][n][z]
  __bf16* Kb   = (__bf16*)carve(xN * 2);
  __bf16* QT   = (__bf16*)carve(xN * 2);        // [bh][z][n]
  __bf16* KT   = (__bf16*)carve(xN * 2);
  __bf16* M1   = (__bf16*)carve(xN * 2);        // [(b,n)][(h,z)]
  __bf16* M2   = (__bf16*)carve(xN * 2);
  float*  rmax = (float*)carve((size_t)BB * HH * NN * 4);
  float*  rsum = (float*)carve((size_t)BB * HH * NN * 4);

  cvt_bf16<<<(int)((xN + 255) / 256), 256, 0, stream>>>(x, xb, (int)xN);
  cvt_w<<<(int)((wN + 255) / 256), 256, 0, stream>>>(Wq, Wqb, WqT);
  cvt_w<<<(int)((wN + 255) / 256), 256, 0, stream>>>(Wk, Wkb, WkT);

  gemm_proj<<<dim3(MROWS / 64, DIMM / 64), 128, 0, stream>>>(xb, Wqb, Qb, QT);
  gemm_proj<<<dim3(MROWS / 64, DIMM / 64), 128, 0, stream>>>(xb, Wkb, Kb, KT);

  attn_stats<<<dim3(NN / 64, BB * HH), 128, 0, stream>>>(Qb, Kb, betas, rmax, rsum);
  attn_m1<<<dim3(NN / 64, BB * HH), 128, 0, stream>>>(Qb, Kb, KT, betas, rmax, rsum, M1);
  attn_m2<<<dim3(NN / 64, BB * HH), 128, 0, stream>>>(Qb, Kb, QT, betas, rmax, rsum, M2);

  gemm_out<<<dim3(MROWS / 64, DIMM / 64), 128, 0, stream>>>(M1, M2, WqT, WkT, out);
}
